// MultiCELoss_30485677867131
// MI455X (gfx1250) — matrix operand order repaired and checked by reading, not compiled, not run on hardware
//
#include <hip/hip_runtime.h>
#include <hip/hip_bf16.h>

#define ALPHA_C 0.2f
#define BETA_C  1.0f
#define EPS_C   1e-8f

constexpr int Bn  = 4096;
constexpr int Rr  = 18;
constexpr int Dd  = 256;
constexpr int RM1 = 17;
constexpr int Nn  = Bn * RM1;   // 69632 targets

typedef _Float16 v16h __attribute__((ext_vector_type(16)));
typedef _Float16 v8h  __attribute__((ext_vector_type(8)));
typedef float    v8f  __attribute__((ext_vector_type(8)));

// ---------------------------------------------------------------------------
// Pass 1: split fp32 features into (hi,lo) f16 pairs, targets in column order.
// col j < B       -> feature row j*R + 1              (pos)
// col j >= B      -> feature row ((j-B)/16)*R + 2 + (j-B)%16   (negs)
// anchors b       -> feature row b*R
// ---------------------------------------------------------------------------
__global__ void split_kernel(const float* __restrict__ feat,
                             _Float16* __restrict__ thi, _Float16* __restrict__ tlo,
                             _Float16* __restrict__ ahi, _Float16* __restrict__ alo)
{
    int rid = blockIdx.x;       // 0 .. Nn+Bn-1
    int d   = threadIdx.x;      // 0 .. 255
    if (rid < Nn) {
        int src = (rid < Bn) ? (rid * Rr + 1)
                             : (((rid - Bn) >> 4) * Rr + 2 + ((rid - Bn) & 15));
        float x = feat[(size_t)src * Dd + d];
        _Float16 h = (_Float16)x;
        _Float16 l = (_Float16)(x - (float)h);
        thi[(size_t)rid * Dd + d] = h;
        tlo[(size_t)rid * Dd + d] = l;
    } else {
        int b = rid - Nn;
        float x = feat[(size_t)b * Rr * Dd + d];
        _Float16 h = (_Float16)x;
        _Float16 l = (_Float16)(x - (float)h);
        ahi[(size_t)b * Dd + d] = h;
        alo[(size_t)b * Dd + d] = l;
    }
}

// ---------------------------------------------------------------------------
// Pass 2: KL term (cosine softmax vs score softmax) + diagonal logits.
// One block (128 thr / 4 waves) per row b.
// ---------------------------------------------------------------------------
__global__ void kl_diag_kernel(const float* __restrict__ feat,
                               const float* __restrict__ scores,
                               const float* __restrict__ lsc,
                               float* __restrict__ klb,
                               float* __restrict__ diag)
{
    __shared__ float sA[Dd];
    __shared__ float sred[128];
    __shared__ float sAn;
    __shared__ float cosv[RM1];

    int b    = blockIdx.x;
    int tid  = threadIdx.x;
    int lane = tid & 31;
    int w    = tid >> 5;
    size_t base = (size_t)b * Rr * Dd;

    float p = 0.f;
    for (int d = tid; d < Dd; d += 128) { float x = feat[base + d]; sA[d] = x; p += x * x; }
    sred[tid] = p;
    __syncthreads();
    if (tid == 0) {
        float s = 0.f;
        for (int i = 0; i < 128; ++i) s += sred[i];
        sAn = fmaxf(sqrtf(s), EPS_C);
    }
    __syncthreads();

    float sc = *lsc;
    for (int k = w; k < RM1; k += 4) {
        const float* tp = feat + base + (size_t)(k + 1) * Dd;
        float dp = 0.f, q = 0.f;
        for (int d = lane; d < Dd; d += 32) { float t = tp[d]; dp += sA[d] * t; q += t * t; }
        for (int off = 16; off > 0; off >>= 1) { dp += __shfl_xor(dp, off); q += __shfl_xor(q, off); }
        if (lane == 0) {
            float tn = fmaxf(sqrtf(q), EPS_C);
            cosv[k] = dp / (sAn * tn);
            if (k == 0) diag[b] = sc * dp;   // logits[b,b] = sc * <anchor, pos>
        }
    }
    __syncthreads();

    if (tid == 0) {
        float mx = -__builtin_inff();
        for (int k = 0; k < RM1; ++k) mx = fmaxf(mx, cosv[k]);
        float se = 0.f;
        for (int k = 0; k < RM1; ++k) se += __expf(cosv[k] - mx);
        float lse = mx + __logf(se);

        const float* srow = scores + (size_t)b * RM1;
        float smx = -__builtin_inff();
        for (int k = 0; k < RM1; ++k) smx = fmaxf(smx, srow[k]);
        float ss = 0.f;
        for (int k = 0; k < RM1; ++k) ss += __expf(srow[k] - smx);
        float lss = __logf(ss);

        float kl = 0.f;
        for (int k = 0; k < RM1; ++k) {
            float lc = srow[k] - smx - lss;   // log ce_k
            float ce = __expf(lc);
            float el = cosv[k] - lse;         // emb_log_k
            kl += ce * (lc - el);
        }
        klb[b] = kl / (float)RM1;
    }
}

// ---------------------------------------------------------------------------
// Pass 3: WMMA GEMM (M=4096, N=69632, K=256) with split-f16 operands and
// online logsumexp. Block = 8 waves; block owns one 16-row M tile and one
// 4096-column N chunk; wave owns 32 16-col C tiles.
// ---------------------------------------------------------------------------
__global__ __launch_bounds__(256)
void gemm_lse_kernel(const _Float16* __restrict__ thi, const _Float16* __restrict__ tlo,
                     const _Float16* __restrict__ ahi, const _Float16* __restrict__ alo,
                     const float* __restrict__ lsc,
                     float* __restrict__ pmax, float* __restrict__ psum)
{
    const int tid   = threadIdx.x;
    const int lane  = tid & 31;
    const int w     = tid >> 5;
    const int m0    = blockIdx.x * 16;   // anchor rows
    const int j0    = blockIdx.y * 4096; // target columns
    const float sc  = *lsc;

    const int rsel  = lane & 15;
    const int abase = (lane < 16) ? 0 : 16;   // A: K {a..a+15} contiguous
    const int bbase = (lane < 16) ? 0 : 16;   // B: K {b..b+15} contiguous

    // A tile (16 anchors x 256 K) resident in registers: 8 chunks x (hi,lo)
    v16h Ahi[8], Alo[8];
    {
        const _Float16* arh = ahi + (size_t)(m0 + rsel) * Dd + abase;
        const _Float16* arl = alo + (size_t)(m0 + rsel) * Dd + abase;
#pragma unroll
        for (int c = 0; c < 8; ++c) {
            v8h h0 = *(const v8h*)(arh + c * 32);
            v8h h1 = *(const v8h*)(arh + c * 32 + 8);
            v8h l0 = *(const v8h*)(arl + c * 32);
            v8h l1 = *(const v8h*)(arl + c * 32 + 8);
            Ahi[c] = __builtin_shufflevector(h0, h1, 0,1,2,3,4,5,6,7,8,9,10,11,12,13,14,15);
            Alo[c] = __builtin_shufflevector(l0, l1, 0,1,2,3,4,5,6,7,8,9,10,11,12,13,14,15);
        }
    }

    float rm[8], rs[8];
#pragma unroll
    for (int s = 0; s < 8; ++s) { rm[s] = -__builtin_inff(); rs[s] = 0.f; }

    for (int t = w; t < 256; t += 8) {
        const int j = j0 + t * 16 + rsel;
        const _Float16* brh = thi + (size_t)j * Dd;
        const _Float16* brl = tlo + (size_t)j * Dd;
        if (t + 8 < 256) {   // prefetch next tile's rows (global_prefetch_b8)
            __builtin_prefetch(thi + (size_t)(j + 128) * Dd, 0, 1);
            __builtin_prefetch(tlo + (size_t)(j + 128) * Dd, 0, 1);
        }
        v8f acc = {};
#pragma unroll
        for (int c = 0; c < 8; ++c) {
            v16h Bhi = *(const v16h*)(brh + c * 32 + bbase);
            v16h Blo = *(const v16h*)(brl + c * 32 + bbase);
            acc = __builtin_amdgcn_wmma_f32_16x16x32_f16(false, Ahi[c], false, Bhi,
                                                         (short)0, acc, false, false);
            acc = __builtin_amdgcn_wmma_f32_16x16x32_f16(false, Ahi[c], false, Blo,
                                                         (short)0, acc, false, false);
            acc = __builtin_amdgcn_wmma_f32_16x16x32_f16(false, Alo[c], false, Bhi,
                                                         (short)0, acc, false, false);
        }
        // online logsumexp update per slot (slot s: row s / s+8, this lane's col)
#pragma unroll
        for (int s = 0; s < 8; ++s) {
            float v  = sc * acc[s];
            float nm = fmaxf(rm[s], v);
            rs[s] = rs[s] * __expf(rm[s] - nm) + __expf(v - nm);
            rm[s] = nm;
        }
    }

    // reduce the 16 lanes that share a row (lanes 0-15: rows 0-7; 16-31: rows 8-15)
#pragma unroll
    for (int s = 0; s < 8; ++s) {
        float m = rm[s], su = rs[s];
#pragma unroll
        for (int k = 1; k < 16; k <<= 1) {
            float om = __shfl_xor(m, k);
            float os = __shfl_xor(su, k);
            float nm = fmaxf(m, om);
            su = su * __expf(m - nm) + os * __expf(om - nm);
            m  = nm;
        }
        rm[s] = m; rs[s] = su;
    }

    __shared__ float redM[8][16];
    __shared__ float redS[8][16];
    if (lane == 0 || lane == 16) {
        int roff = (lane == 0) ? 0 : 8;
#pragma unroll
        for (int s = 0; s < 8; ++s) { redM[w][roff + s] = rm[s]; redS[w][roff + s] = rs[s]; }
    }
    __syncthreads();
    if (tid < 16) {
        float m = redM[0][tid], su = redS[0][tid];
        for (int ww = 1; ww < 8; ++ww) {
            float om = redM[ww][tid], os = redS[ww][tid];
            float nm = fmaxf(m, om);
            su = su * __expf(m - nm) + os * __expf(om - nm);
            m  = nm;
        }
        pmax[(size_t)blockIdx.y * Bn + m0 + tid] = m;
        psum[(size_t)blockIdx.y * Bn + m0 + tid] = su;
    }
}

// ---------------------------------------------------------------------------
// Pass 4: combine chunk partials -> lse per row; final scalar loss.
// ---------------------------------------------------------------------------
__global__ void final_kernel(const float* __restrict__ pmax, const float* __restrict__ psum,
                             const float* __restrict__ klb, const float* __restrict__ diag,
                             float* __restrict__ out)
{
    __shared__ float sInf[256];
    __shared__ float sKl[256];
    int tid = threadIdx.x;
    float accI = 0.f, accK = 0.f;
    for (int b = tid; b < Bn; b += 256) {
        float m = -__builtin_inff(), s = 0.f;
        for (int c = 0; c < RM1; ++c) {
            float cm = pmax[(size_t)c * Bn + b];
            float cs = psum[(size_t)c * Bn + b];
            float nm = fmaxf(m, cm);
            s = s * __expf(m - nm) + cs * __expf(cm - nm);
            m = nm;
        }
        float lse = m + __logf(s);
        accI += lse - diag[b];
        accK += klb[b];
    }
    sInf[tid] = accI; sKl[tid] = accK;
    __syncthreads();
    if (tid == 0) {
        float si = 0.f, sk = 0.f;
        for (int i = 0; i < 256; ++i) { si += sInf[i]; sk += sKl[i]; }
        out[0] = BETA_C * sk / (float)Bn + ALPHA_C * si / (float)Bn;
    }
}

// ---------------------------------------------------------------------------
extern "C" void kernel_launch(void* const* d_in, const int* in_sizes, int n_in,
                              void* d_out, int out_size, void* d_ws, size_t ws_size,
                              hipStream_t stream) {
    const float* feat   = (const float*)d_in[0];
    const float* scores = (const float*)d_in[1];
    const float* lsc    = (const float*)d_in[3];
    (void)in_sizes; (void)n_in; (void)out_size; (void)ws_size;

    // workspace carve (~76 MB)
    char* p = (char*)d_ws;
    auto carve = [&](size_t bytes) { char* r = p; p += (bytes + 255) & ~(size_t)255; return r; };
    _Float16* thi  = (_Float16*)carve((size_t)Nn * Dd * sizeof(_Float16));
    _Float16* tlo  = (_Float16*)carve((size_t)Nn * Dd * sizeof(_Float16));
    _Float16* ahi  = (_Float16*)carve((size_t)Bn * Dd * sizeof(_Float16));
    _Float16* alo  = (_Float16*)carve((size_t)Bn * Dd * sizeof(_Float16));
    float*    pmax = (float*)carve((size_t)RM1 * Bn * sizeof(float));
    float*    psum = (float*)carve((size_t)RM1 * Bn * sizeof(float));
    float*    klb  = (float*)carve((size_t)Bn * sizeof(float));
    float*    diag = (float*)carve((size_t)Bn * sizeof(float));

    split_kernel  <<<Nn + Bn, Dd, 0, stream>>>(feat, thi, tlo, ahi, alo);
    kl_diag_kernel<<<Bn, 128, 0, stream>>>(feat, scores, lsc, klb, diag);
    gemm_lse_kernel<<<dim3(Bn / 16, RM1), 256, 0, stream>>>(thi, tlo, ahi, alo, lsc, pmax, psum);
    final_kernel  <<<1, 256, 0, stream>>>(pmax, psum, klb, diag, (float*)d_out);
}
